// Multi_Head_Attention_33827162423765
// MI455X (gfx1250) — matrix operand order repaired and checked by reading, not compiled, not run on hardware
//
#include <hip/hip_runtime.h>

// ---- problem sizes (from reference setup_inputs) ----
constexpr int B_ = 4;
constexpr int S_ = 2048;
constexpr int D_ = 512;
constexpr int H_ = 8;
constexpr float SCALE_ = 0.04419417382f;   // 1/sqrt(512)
constexpr float NEG_BIG_ = 1.0e10f;

typedef __attribute__((ext_vector_type(16))) __bf16 v16bf;
typedef __attribute__((ext_vector_type(8)))  __bf16 v8bf;
typedef __attribute__((ext_vector_type(2)))  __bf16 v2bf;
typedef __attribute__((ext_vector_type(8)))  float  v8f;
typedef __attribute__((ext_vector_type(4)))  float  f4v;

// native bf16 convert (RNE) — clang emits v_cvt(_pk)_bf16_f32
__device__ __forceinline__ __bf16 f2bf(float f) { return (__bf16)f; }

// load 8 consecutive f32 (16B-aligned) and convert to 8 bf16
__device__ __forceinline__ v8bf cvt8(const float* p) {
    f4v a = *(const f4v*)p;
    f4v b = *(const f4v*)(p + 4);
    v8bf r;
    r[0] = f2bf(a[0]); r[1] = f2bf(a[1]); r[2] = f2bf(a[2]); r[3] = f2bf(a[3]);
    r[4] = f2bf(b[0]); r[5] = f2bf(b[1]); r[6] = f2bf(b[2]); r[7] = f2bf(b[3]);
    return r;
}

__device__ __forceinline__ v16bf comb(v8bf lo, v8bf hi) {
    return __builtin_shufflevector(lo, hi, 0,1,2,3,4,5,6,7,8,9,10,11,12,13,14,15);
}

__device__ __forceinline__ v16bf ld16bf(const __bf16* p) {
    return comb(*(const v8bf*)p, *(const v8bf*)(p + 8));
}

__device__ __forceinline__ v8f wmma_bf16(v16bf a, v16bf b, v8f c) {
    return __builtin_amdgcn_wmma_f32_16x16x32_bf16(
        /*neg_a=*/false, a, /*neg_b=*/false, b,
        /*c_mod=*/(short)0, c, /*reuse_a=*/false, /*reuse_b=*/false);
}

// =====================================================================
// Kernel 1: Q/K/V projections.  Y[b,h] = X[b] (SxD) @ W[h] (DxD)
//  grid.x = (S/128)*(D/128), grid.y = h*3+w, grid.z = b; 256 thr = 8 waves.
//  Each wave: 16 rows x 128 cols (8 WMMA accumulators) — one A-fragment
//  feeds 8 WMMAs per k-step.
//  W tile (32k x 128n) double-buffered in LDS, transposed via k-pair packing
//  (v_cvt_pk_bf16_f32 -> single ds_store_b32 per column pair).
//  w==0 -> Qb row-major, w==1 -> Kb row-major, w==2 -> Vt transposed [d][t].
// =====================================================================
__global__ __launch_bounds__(256) void proj_kernel(
    const float* __restrict__ X,
    const float* __restrict__ W0, const float* __restrict__ W1,
    const float* __restrict__ W2,
    __bf16* __restrict__ Qb, __bf16* __restrict__ Kb, __bf16* __restrict__ Vt)
{
    __shared__ __bf16 Wt[2][128][40];      // [buf][n][k], 80B rows (16B aligned)

    const int b  = blockIdx.z;
    const int h  = blockIdx.y / 3;
    const int w  = blockIdx.y % 3;
    const int nTiles = D_ / 128;           // 4
    const int nT = blockIdx.x % nTiles;
    const int rT = blockIdx.x / nTiles;
    const int n0 = nT * 128;

    const int tid  = threadIdx.x;
    const int wv   = tid >> 5;
    const int lane = tid & 31;
    const int lm   = lane & 15;
    const int lg   = lane >> 4;
    const int r0   = rT * 128 + wv * 16;

    const float* W  = (w == 0) ? W0 : ((w == 1) ? W1 : W2);
    const float* Wh = W + (size_t)h * D_ * D_;
    const float* Xb = X + (size_t)b * S_ * D_;

    v8f acc[8];
#pragma unroll
    for (int i = 0; i < 8; ++i) acc[i] = (v8f)0.0f;

    // staging: 32k x 128n tile = 512 (k-pair, n4) units; 2 units per thread.
    //  unit -> n4 = (unit%32)*4, k2 = (unit/32)*2 ; consecutive tid -> same
    //  k-row, consecutive 16B chunks (coalesced). Transposed write packs the
    //  (k2,k2+1) pair for each column into one b32 LDS store.
    auto stage = [&](int k0, __bf16 (*buf)[40]) {
#pragma unroll
        for (int u = 0; u < 2; ++u) {
            int unit = tid + u * 256;
            int n4 = (unit & 31) * 4;
            int k2 = (unit >> 5) * 2;
            const float* wp = Wh + (size_t)(k0 + k2) * D_ + n0 + n4;
            f4v r0v = *(const f4v*)wp;
            f4v r1v = *(const f4v*)(wp + D_);
#pragma unroll
            for (int j = 0; j < 4; ++j) {
                v2bf p;
                p[0] = f2bf(r0v[j]);
                p[1] = f2bf(r1v[j]);
                *(v2bf*)&buf[n4 + j][k2] = p;
            }
        }
    };

    stage(0, Wt[0]);

    const int NSTEP = D_ / 32;             // 16
    for (int step = 0; step < NSTEP; ++step) {
        __syncthreads();                   // staging of `step` visible; prev reads done
        const int cur = step & 1;
        if (step + 1 < NSTEP) stage((step + 1) * 32, Wt[cur ^ 1]);

        // A fragment: rows of X (f32 -> bf16), two 8-wide k-runs per lane
        const float* xr = Xb + (size_t)(r0 + lm) * D_ + step * 32 + 8 * lg;
        v16bf af = comb(cvt8(xr), cvt8(xr + 16));

        // batch the B-fragment LDS loads, then the WMMA chain
        v16bf bf[8];
#pragma unroll
        for (int nt = 0; nt < 8; ++nt)
            bf[nt] = comb(*(const v8bf*)&Wt[cur][nt * 16 + lm][8 * lg], *(const v8bf*)&Wt[cur][nt * 16 + lm][8 * lg + 16]);
#pragma unroll
        for (int nt = 0; nt < 8; ++nt)
            acc[nt] = wmma_bf16(af, bf[nt], acc[nt]);
    }

    if (w < 2) {
        __bf16* Y = ((w == 0) ? Qb : Kb) + (size_t)(b * H_ + h) * S_ * D_;
#pragma unroll
        for (int nt = 0; nt < 8; ++nt) {
#pragma unroll
            for (int e = 0; e < 8; ++e) {
                int m = e + 8 * lg;
                Y[(size_t)(r0 + m) * D_ + n0 + nt * 16 + lm] = f2bf(acc[nt][e]);
            }
        }
    } else {
        // V stored transposed: Vt[b,h][d][t]; accumulator rows are contiguous
        __bf16* Y = Vt + (size_t)(b * H_ + h) * D_ * S_;
#pragma unroll
        for (int nt = 0; nt < 8; ++nt) {
            int col = n0 + nt * 16 + lm;
            v8bf pk;
#pragma unroll
            for (int e = 0; e < 8; ++e) pk[e] = f2bf(acc[nt][e]);
            *(v8bf*)&Y[(size_t)col * S_ + r0 + 8 * lg] = pk;
        }
    }
}

// =====================================================================
// Kernel 2: attention with head-mean fused.
//  grid.x = S/16 query-row blocks, grid.y = b. 128 thr = 4 waves.
//  16x2048 f32 score panel in LDS (131KB) -> 2 workgroups per WGP.
//  Per head: A) scores via WMMA (waves split columns 4-way),
//            B) row softmax in LDS (float4 sweeps),
//            C) P@V^T via WMMA (waves split d-range 4-way, 8 accums).
// =====================================================================
#define PSTRIDE 2052   // 2048 + 4 floats pad (16B alignment kept, banks spread)

__global__ __launch_bounds__(128) void attn_kernel(
    const __bf16* __restrict__ Qb, const __bf16* __restrict__ Kb,
    const __bf16* __restrict__ Vt, const float* __restrict__ mask,
    float* __restrict__ out)
{
    extern __shared__ float panel[];                 // 16*PSTRIDE + 256
    float* scratchM = panel + 16 * PSTRIDE;          // 128 floats
    float* scratchS = scratchM + 128;                // 128 floats

    const int b  = blockIdx.y;
    const int q0 = blockIdx.x * 16;

    const int tid  = threadIdx.x;
    const int wv   = tid >> 5;
    const int lane = tid & 31;
    const int lm   = lane & 15;
    const int lg   = lane >> 4;

    v8f oacc[8];
#pragma unroll
    for (int i = 0; i < 8; ++i) oacc[i] = (v8f)0.0f;

    for (int h = 0; h < H_; ++h) {
        const __bf16* Qh = Qb + (size_t)(b * H_ + h) * S_ * D_;
        const __bf16* Kh = Kb + (size_t)(b * H_ + h) * S_ * D_;
        const __bf16* Vh = Vt + (size_t)(b * H_ + h) * D_ * S_;

        // ---- preload the block's 16 Q rows as 16 resident A-fragments ----
        v16bf qf[16];
        {
            const __bf16* qp = Qh + (size_t)(q0 + lm) * D_;
#pragma unroll
            for (int c = 0; c < 16; ++c) {
                const __bf16* p = qp + c * 32 + 16 * lg;
                qf[c] = comb(*(const v8bf*)p, *(const v8bf*)(p + 8));
            }
        }

        // ---- Phase A: scores = scale * Q K^T, masked -> LDS panel ----
        for (int ct = wv; ct < S_ / 16; ct += 4) {
            v8f sa = (v8f)0.0f;
            const int t0 = ct * 16;
            const __bf16* kp = Kh + (size_t)(t0 + lm) * D_ + 16 * lg;
            // double-buffer the K fragment against the WMMA chain
            v16bf bf = ld16bf(kp);
#pragma unroll
            for (int c = 0; c < 16; ++c) {
                v16bf nb;
                if (c < 15) nb = ld16bf(kp + (c + 1) * 32);
                sa = wmma_bf16(qf[c], bf, sa);
                bf = nb;
            }
            const int col = t0 + lm;
#pragma unroll
            for (int e = 0; e < 8; ++e) {
                int row = e + 8 * lg;
                float mv = mask[((size_t)b * S_ + (q0 + row)) * S_ + col];
                float s  = sa[e] * SCALE_;
                panel[row * PSTRIDE + col] = s * mv - NEG_BIG_ * (1.0f - mv);
            }
        }
        __syncthreads();

        // ---- Phase B: row softmax (thread -> row tid%16, 256-col segment) ----
        {
            const int row = tid & 15;
            const int seg = tid >> 4;                // 8 segments x 256 cols
            float* pr = panel + row * PSTRIDE + seg * 256;
            float mx = -3.4e38f;
            for (int i = 0; i < 64; ++i) {
                f4v v = *(const f4v*)(pr + i * 4);
                mx = fmaxf(mx, fmaxf(fmaxf(v[0], v[1]), fmaxf(v[2], v[3])));
            }
            scratchM[row * 8 + seg] = mx;
            __syncthreads();
            float rmax = scratchM[row * 8];
#pragma unroll
            for (int k = 1; k < 8; ++k) rmax = fmaxf(rmax, scratchM[row * 8 + k]);
            float sum = 0.0f;
            for (int i = 0; i < 64; ++i) {
                f4v v = *(f4v*)(pr + i * 4);
                f4v e;
                e[0] = __expf(v[0] - rmax); e[1] = __expf(v[1] - rmax);
                e[2] = __expf(v[2] - rmax); e[3] = __expf(v[3] - rmax);
                *(f4v*)(pr + i * 4) = e;
                sum += (e[0] + e[1]) + (e[2] + e[3]);
            }
            scratchS[row * 8 + seg] = sum;
            __syncthreads();
            float rsum = scratchS[row * 8];
#pragma unroll
            for (int k = 1; k < 8; ++k) rsum += scratchS[row * 8 + k];
            float inv = 1.0f / rsum;
            for (int i = 0; i < 64; ++i) {
                f4v v = *(f4v*)(pr + i * 4);
                v[0] *= inv; v[1] *= inv; v[2] *= inv; v[3] *= inv;
                *(f4v*)(pr + i * 4) = v;
            }
        }
        __syncthreads();

        // ---- Phase C: O += P @ V^T ; wave covers 16 rows x 128 d-cols ----
        {
            const int d0 = wv * 128;
            const float* prow = panel + lm * PSTRIDE;
            for (int c = 0; c < S_ / 32; ++c) {
                const float* pp = prow + c * 32 + 16 * lg;
                v16bf af = comb(cvt8(pp), cvt8(pp + 8));
                // batch the 8 V-fragment global loads, then the WMMA chain
                v16bf bf[8];
#pragma unroll
                for (int nt = 0; nt < 8; ++nt)
                    bf[nt] = ld16bf(Vh + (size_t)(d0 + nt * 16 + lm) * S_
                                       + c * 32 + 16 * lg);
#pragma unroll
                for (int nt = 0; nt < 8; ++nt)
                    oacc[nt] = wmma_bf16(af, bf[nt], oacc[nt]);
            }
        }
        __syncthreads();   // panel reused next head
    }

    // ---- head mean -> d_out (f32, row-major B,S,D) ----
    const float invH = 1.0f / (float)H_;
    const int d0 = wv * 128;
#pragma unroll
    for (int nt = 0; nt < 8; ++nt) {
#pragma unroll
        for (int e = 0; e < 8; ++e) {
            int row = q0 + e + 8 * lg;
            int col = d0 + nt * 16 + lm;
            out[((size_t)b * S_ + row) * D_ + col] = oacc[nt][e] * invH;
        }
    }
}

// =====================================================================
extern "C" void kernel_launch(void* const* d_in, const int* in_sizes, int n_in,
                              void* d_out, int out_size, void* d_ws, size_t ws_size,
                              hipStream_t stream) {
    const float* X    = (const float*)d_in[0];
    const float* mask = (const float*)d_in[1];
    const float* Wq   = (const float*)d_in[2];
    const float* Wk   = (const float*)d_in[3];
    const float* Wv   = (const float*)d_in[4];
    float* out = (float*)d_out;

    const size_t QELEMS = (size_t)B_ * H_ * S_ * D_;        // 33,554,432
    char* wsb = (char*)d_ws;
    __bf16* Qb = (__bf16*)(wsb);
    __bf16* Kb = (__bf16*)(wsb + QELEMS * 2);
    __bf16* Vt = (__bf16*)(wsb + QELEMS * 4);

    // QKV projections: grid (S/128 * D/128, H*3, B)
    proj_kernel<<<dim3((S_ / 128) * (D_ / 128), H_ * 3, B_), 256, 0, stream>>>(
        X, Wq, Wk, Wv, Qb, Kb, Vt);

    // attention + head mean: grid (S/16, B), ~131KB dynamic LDS score panel
    size_t shbytes = (size_t)(16 * PSTRIDE + 256) * sizeof(float);
    attn_kernel<<<dim3(S_ / 16, B_), 128, shbytes, stream>>>(
        Qb, Kb, Vt, mask, out);
}
